// RoPEAttention_90778428769132
// MI455X (gfx1250) — matrix operand order repaired and checked by reading, not compiled, not run on hardware
//
#include <hip/hip_runtime.h>
#include <hip/hip_bf16.h>

// ---------------------------------------------------------------------------
// Types for CDNA5 WMMA (wave32): D(16x16 f32) = A(16x32 bf16) x B(32x16 bf16) + C
// ---------------------------------------------------------------------------
typedef __attribute__((ext_vector_type(16))) __bf16 v16bf;
typedef __attribute__((ext_vector_type(8)))  __bf16 v8bf;
typedef __attribute__((ext_vector_type(8)))  float  v8f;
typedef int v4i_ __attribute__((vector_size(16)));   // matches async-LDS builtin param

static __device__ __forceinline__ v16bf cmb(v8bf lo, v8bf hi) {
    return __builtin_shufflevector(lo, hi, 0,1,2,3,4,5,6,7,8,9,10,11,12,13,14,15);
}
static __device__ __forceinline__ v8bf ld8(const __bf16* p) {
    return *(const v8bf*)p;
}
static __device__ __forceinline__ v8f wmma_bf16(v16bf a, v16bf b, v8f c) {
    return __builtin_amdgcn_wmma_f32_16x16x32_bf16(false, a, false, b, (short)0, c, false, false);
}

#define BB 8
#define NN 1024
#define CC 1024
#define HH 16
#define DD 64

// ---------------------------------------------------------------------------
// CDNA5 async global->LDS copy (ASYNCcnt path) with portable fallback.
// Probe result: param0 is AS1 (global) v4i*, param1 is LDS pointer.
// ---------------------------------------------------------------------------
#if defined(__has_builtin)
#if __has_builtin(__builtin_amdgcn_global_load_async_to_lds_b128)
#define HAVE_ASYNC_LDS 1
#endif
#endif

static __device__ __forceinline__ void copy16_g2l(const __bf16* g, __bf16* l) {
#if defined(HAVE_ASYNC_LDS)
    __builtin_amdgcn_global_load_async_to_lds_b128(
        (__attribute__((address_space(1))) v4i_*)g,
        (__attribute__((address_space(3))) v4i_*)l,
        0, 0);
#else
    *(v8bf*)l = *(const v8bf*)g;
#endif
}

static __device__ __forceinline__ void async_wait0() {
#if defined(HAVE_ASYNC_LDS)
#if __has_builtin(__builtin_amdgcn_s_wait_asynccnt)
    __builtin_amdgcn_s_wait_asynccnt(0);
#else
    asm volatile("s_wait_asynccnt 0x0" ::: "memory");
#endif
#endif
}

// ---------------------------------------------------------------------------
// Kernel 1: fp32 -> bf16 cast (row-major preserved)
// ---------------------------------------------------------------------------
__global__ void cast_bf16_kernel(const float* __restrict__ x, __bf16* __restrict__ y, int n) {
    int i = blockIdx.x * blockDim.x + threadIdx.x;
    if (i < n) y[i] = (__bf16)x[i];
}

// ---------------------------------------------------------------------------
// Kernel 2: transpose + cast: w (K x Ncol, row-major) -> wt (Ncol x K, row-major)
// ---------------------------------------------------------------------------
__global__ void tcast_bf16_kernel(const float* __restrict__ w, __bf16* __restrict__ wt,
                                  int K, int Ncol) {
    int idx = blockIdx.x * blockDim.x + threadIdx.x;
    if (idx >= K * Ncol) return;
    int n = idx / K;
    int k = idx - n * K;
    wt[idx] = (__bf16)w[(size_t)k * Ncol + n];
}

// ---------------------------------------------------------------------------
// Kernel 3 / 6: WMMA GEMM.  C(MxNc,f32) = A(MxKd bf16 row-major) x Bt^T,
// Bt stored (Nc x Kd) row-major ([n][k] layout -> contiguous B fragments).
// Block = 256 threads (8 waves), tile 128(M) x 128(N), K-step 32.
// Double-buffered LDS staging via async global->LDS copies (ASYNCcnt).
// Wave tile: 32(M) x 64(N) = 8 WMMAs per K-step per wave.
// ---------------------------------------------------------------------------
#define GSTR 40   // 32 + 8 pad elements; 80-byte rows (16B-aligned chunks)

__global__ __launch_bounds__(256) void gemm_wmma_bf16(
    const __bf16* __restrict__ A, const __bf16* __restrict__ Bt,
    float* __restrict__ Cout, const float* __restrict__ bias,
    int M, int Nc, int Kd)
{
    __shared__ __bf16 Al[2][128 * GSTR];   // 2 x 10 KB
    __shared__ __bf16 Bl[2][128 * GSTR];   // 2 x 10 KB

    const int nBase = blockIdx.x * 128;
    const int mBase = blockIdx.y * 128;
    const int tid   = threadIdx.x;
    const int lane  = tid & 31;
    const int w     = tid >> 5;         // wave 0..7
    const int wm    = w & 3;            // M subtile (32 rows each)
    const int wn    = w >> 2;           // N subtile (64 cols each)
    const int lm    = lane & 15;
    const int lh    = lane >> 4;

    v8f acc[2][4];
    #pragma unroll
    for (int s = 0; s < 2; ++s)
        #pragma unroll
        for (int j = 0; j < 4; ++j)
            acc[s][j] = (v8f){0.f,0.f,0.f,0.f,0.f,0.f,0.f,0.f};

    // stage one 128x32 A tile + 128x32 B tile: 1024 16B chunks, 4 per thread
    auto stage_tile = [&](int sb, int kk2) {
        #pragma unroll
        for (int c = tid; c < 512; c += 256) {
            int r  = c >> 2;
            int ch = (c & 3) * 8;
            copy16_g2l(&A[(size_t)(mBase + r) * Kd + kk2 + ch], &Al[sb][r * GSTR + ch]);
            copy16_g2l(&Bt[(size_t)(nBase + r) * Kd + kk2 + ch], &Bl[sb][r * GSTR + ch]);
        }
    };

    stage_tile(0, 0);
    int stage = 0;

    for (int kk = 0; kk < Kd; kk += 32) {
        async_wait0();
        __syncthreads();                 // buf[stage] visible to all waves;
                                         // everyone done reading buf[stage^1]
        int nxt = kk + 32;
        if (nxt < Kd) stage_tile(stage ^ 1, nxt);

        // A fragments: rows wm*32 + s*16 + lm, k 16*lh .. 16*lh+15
        v16bf af[2];
        #pragma unroll
        for (int s = 0; s < 2; ++s) {
            const __bf16* ar = &Al[stage][(wm * 32 + s * 16 + lm) * GSTR];
            af[s] = cmb(ld8(ar + 16 * lh), ld8(ar + 16 * lh + 8));
        }
        #pragma unroll
        for (int j = 0; j < 4; ++j) {
            const __bf16* br = &Bl[stage][(wn * 64 + j * 16 + lm) * GSTR + 16 * lh];
            v16bf bf_ = cmb(ld8(br), ld8(br + 8));
            acc[0][j] = wmma_bf16(af[0], bf_, acc[0][j]);
            acc[1][j] = wmma_bf16(af[1], bf_, acc[1][j]);
        }
        stage ^= 1;
    }

    // Epilogue: D layout: lane lm = col-in-tile, row = r + 8*lh
    #pragma unroll
    for (int s = 0; s < 2; ++s) {
        #pragma unroll
        for (int j = 0; j < 4; ++j) {
            int col = nBase + wn * 64 + j * 16 + lm;
            float bv = bias ? bias[col] : 0.f;
            #pragma unroll
            for (int r = 0; r < 8; ++r) {
                int row = mBase + wm * 32 + s * 16 + r + 8 * lh;
                Cout[(size_t)row * Nc + col] = acc[s][j][r] + bv;
            }
        }
    }
}

// ---------------------------------------------------------------------------
// Kernel 4: RoPE + pack q/k/v.
// qkv fp32 is (B*N, 3*C), col = which*1024 + h*64 + d.
// Q/K out: bf16 [b][h][n][d] (Q pre-scaled by 1/sqrt(64)=0.125)
// V  out: bf16 [b][h][d][n]  (transposed so P@V B-operand is k-contiguous)
// ---------------------------------------------------------------------------
__global__ void rope_pack_kernel(const float* __restrict__ qkv,
                                 __bf16* __restrict__ Q,
                                 __bf16* __restrict__ Kb,
                                 __bf16* __restrict__ Vt)
{
    int idx = blockIdx.x * blockDim.x + threadIdx.x;     // B*N*H*32
    if (idx >= BB * NN * HH * 32) return;
    int i = idx & 31;
    int t = idx >> 5;
    int h = t & (HH - 1);   t >>= 4;
    int n = t & (NN - 1);
    int b = t >> 10;

    const float* src = qkv + (size_t)(b * NN + n) * (3 * CC);
    int d0 = h * DD + 2 * i;
    float q1 = src[d0],           q2 = src[d0 + 1];
    float k1 = src[CC + d0],      k2 = src[CC + d0 + 1];
    float v1 = src[2 * CC + d0],  v2 = src[2 * CC + d0 + 1];

    float omega = __powf(10000.0f, -(float)i * (1.0f / 32.0f));
    float fr = (float)n * omega;
    float s, c;
    __sincosf(fr, &s, &c);

    size_t qb = (((size_t)(b * HH + h)) * NN + n) * DD + 2 * i;
    Q[qb]     = (__bf16)((q1 * c - q2 * s) * 0.125f);
    Q[qb + 1] = (__bf16)((q2 * c + q1 * s) * 0.125f);
    Kb[qb]     = (__bf16)(k1 * c - k2 * s);
    Kb[qb + 1] = (__bf16)(k2 * c + k1 * s);

    size_t vb = (((size_t)(b * HH + h)) * DD + 2 * i) * NN + n;
    Vt[vb]      = (__bf16)v1;
    Vt[vb + NN] = (__bf16)v2;
}

// ---------------------------------------------------------------------------
// Kernel 5: flash attention, one wave32 per (b, h, 16-query tile).
// ---------------------------------------------------------------------------
__global__ __launch_bounds__(32) void attn_wmma_kernel(
    const __bf16* __restrict__ Q, const __bf16* __restrict__ Km,
    const __bf16* __restrict__ Vt, __bf16* __restrict__ Ao)
{
    __shared__ __bf16 P[16 * 32];  // D-fragment -> A-fragment layout bridge

    const int lane = threadIdx.x;
    const int lm = lane & 15;
    const int lh = lane >> 4;
    const int qt = blockIdx.x;
    const int h  = blockIdx.y;
    const int b  = blockIdx.z;
    const size_t bh = (size_t)(b * HH + h);

    const __bf16* Qp = Q + (bh * NN + qt * 16 + lm) * DD;
    v16bf qa0 = cmb(ld8(Qp +      16 * lh), ld8(Qp +  8 + 16 * lh));  // d 0..31
    v16bf qa1 = cmb(ld8(Qp + 32 + 16 * lh), ld8(Qp + 40 + 16 * lh));  // d 32..63

    const __bf16* Kp = Km + bh * NN * DD;
    const __bf16* Vp = Vt + bh * DD * NN;

    float rm[8], rl[8];
    v8f o[4];
    #pragma unroll
    for (int r = 0; r < 8; ++r) { rm[r] = -3.0e38f; rl[r] = 0.f; }
    #pragma unroll
    for (int t = 0; t < 4; ++t) o[t] = (v8f){0.f,0.f,0.f,0.f,0.f,0.f,0.f,0.f};

    for (int kb = 0; kb < NN; kb += 32) {
        v8f s0 = (v8f){0.f,0.f,0.f,0.f,0.f,0.f,0.f,0.f};
        v8f s1 = (v8f){0.f,0.f,0.f,0.f,0.f,0.f,0.f,0.f};
        {
            const __bf16* kr = Kp + (size_t)(kb + lm) * DD + 16 * lh;
            v16bf b0 = cmb(ld8(kr),      ld8(kr + 8));
            v16bf b1 = cmb(ld8(kr + 32), ld8(kr + 40));
            s0 = wmma_bf16(qa0, b0, s0);
            s0 = wmma_bf16(qa1, b1, s0);
            const __bf16* kr2 = kr + 16 * DD;
            v16bf b2 = cmb(ld8(kr2),      ld8(kr2 + 8));
            v16bf b3 = cmb(ld8(kr2 + 32), ld8(kr2 + 40));
            s1 = wmma_bf16(qa0, b2, s1);
            s1 = wmma_bf16(qa1, b3, s1);
        }

        #pragma unroll
        for (int r = 0; r < 8; ++r) {
            float mx = fmaxf(s0[r], s1[r]);
            #pragma unroll
            for (int sft = 1; sft < 16; sft <<= 1)
                mx = fmaxf(mx, __shfl_xor(mx, sft, 32));
            float mn = fmaxf(rm[r], mx);
            float al = __expf(rm[r] - mn);
            rm[r] = mn;
            float p0 = __expf(s0[r] - mn);
            float p1 = __expf(s1[r] - mn);
            float rs = p0 + p1;
            #pragma unroll
            for (int sft = 1; sft < 16; sft <<= 1)
                rs += __shfl_xor(rs, sft, 32);
            rl[r] = rl[r] * al + rs;
            o[0][r] *= al; o[1][r] *= al; o[2][r] *= al; o[3][r] *= al;
            P[(r + 8 * lh) * 32 + lm]      = (__bf16)p0;
            P[(r + 8 * lh) * 32 + lm + 16] = (__bf16)p1;
        }
        asm volatile("" ::: "memory");   // DS is in-order per wave; fence compiler only

        const __bf16* pr = &P[lm * 32];
        v16bf pa = cmb(*(const v8bf*)(pr + 16 * lh), *(const v8bf*)(pr + 16 * lh + 8));

        #pragma unroll
        for (int t = 0; t < 4; ++t) {
            const __bf16* vr = Vp + (size_t)(t * 16 + lm) * NN + kb + 16 * lh;
            v16bf vb = cmb(ld8(vr), ld8(vr + 8));
            o[t] = wmma_bf16(pa, vb, o[t]);
        }
        asm volatile("" ::: "memory");
    }

    #pragma unroll
    for (int r = 0; r < 8; ++r) {
        float inv = 1.f / rl[r];
        int row = qt * 16 + r + 8 * lh;
        size_t obase = ((size_t)b * NN + row) * CC + h * DD;
        #pragma unroll
        for (int t = 0; t < 4; ++t)
            Ao[obase + t * 16 + lm] = (__bf16)(o[t][r] * inv);
    }
}

// ---------------------------------------------------------------------------
// Launch
// ---------------------------------------------------------------------------
extern "C" void kernel_launch(void* const* d_in, const int* in_sizes, int n_in,
                              void* d_out, int out_size, void* d_ws, size_t ws_size,
                              hipStream_t stream) {
    const float* x      = (const float*)d_in[0];   // (B,N,C)
    const float* w_qkv  = (const float*)d_in[1];   // (C, 3C)
    const float* w_proj = (const float*)d_in[2];   // (C, C)
    const float* b_proj = (const float*)d_in[3];   // (C,)
    float* out = (float*)d_out;                    // (B,N,C) fp32

    char* ws = (char*)d_ws;
    // [0, 96MB): qkv fp32 (8192 x 3072); later reused for attnout bf16 (16MB)
    float*  qkv_f32 = (float*)(ws + 0);
    __bf16* attnout = (__bf16*)(ws + 0);
    __bf16* xbf     = (__bf16*)(ws + 100663296ull);               // 16 MB
    __bf16* wqkvT   = (__bf16*)(ws + 100663296ull + 16777216ull); // 6 MB
    __bf16* wprojT  = (__bf16*)(ws + 100663296ull + 23068672ull); // 2 MB
    __bf16* Qbf     = (__bf16*)(ws + 100663296ull + 25165824ull); // 16 MB
    __bf16* Kbf     = (__bf16*)(ws + 100663296ull + 41943040ull); // 16 MB
    __bf16* Vtbf    = (__bf16*)(ws + 100663296ull + 58720256ull); // 16 MB

    {
        int n = BB * NN * CC;
        cast_bf16_kernel<<<(n + 255) / 256, 256, 0, stream>>>(x, xbf, n);
    }
    {
        int n = CC * 3 * CC;
        tcast_bf16_kernel<<<(n + 255) / 256, 256, 0, stream>>>(w_qkv, wqkvT, CC, 3 * CC);
    }
    {
        int n = CC * CC;
        tcast_bf16_kernel<<<(n + 255) / 256, 256, 0, stream>>>(w_proj, wprojT, CC, CC);
    }

    // QKV GEMM: (8192 x 1024) x (1024 x 3072) -> fp32
    {
        dim3 grid(3 * CC / 128, BB * NN / 128);  // (24, 64)
        gemm_wmma_bf16<<<grid, 256, 0, stream>>>(xbf, wqkvT, qkv_f32, nullptr,
                                                 BB * NN, 3 * CC, CC);
    }

    // RoPE + pack
    {
        int n = BB * NN * HH * 32;
        rope_pack_kernel<<<(n + 255) / 256, 256, 0, stream>>>(qkv_f32, Qbf, Kbf, Vtbf);
    }

    // Flash attention
    {
        dim3 grid(NN / 16, HH, BB);
        attn_wmma_kernel<<<grid, 32, 0, stream>>>(Qbf, Kbf, Vtbf, attnout);
    }

    // Output projection + bias -> d_out
    {
        dim3 grid(CC / 128, BB * NN / 128);      // (8, 64)
        gemm_wmma_bf16<<<grid, 256, 0, stream>>>(attnout, wprojT, out, b_proj,
                                                 BB * NN, CC, CC);
    }
}
